// MultiHeadAttention_33234456937081
// MI455X (gfx1250) — matrix operand order repaired and checked by reading, not compiled, not run on hardware
//
#include <hip/hip_runtime.h>

// ---------------------------------------------------------------------------
// MHA forward for gfx1250 (MI455X): bf16 WMMA + TDM (tensor_load_to_lds)
// double-buffered LDS staging; softmax reductions on VALU (permlane16) and
// XDL (row-sum via P x ones WMMA). B=2, S=2048, H=16, d=64, D_MODEL=1024.
// ---------------------------------------------------------------------------

typedef __attribute__((ext_vector_type(16))) __bf16 v16bf;
typedef __attribute__((ext_vector_type(8)))  __bf16 v8bf;
typedef __attribute__((ext_vector_type(8)))  float  v8f;
typedef __attribute__((ext_vector_type(4)))  unsigned int v4u;
typedef __attribute__((ext_vector_type(8)))  int v8i;
typedef __attribute__((ext_vector_type(4)))  int v4i;

#define SEQ   2048
#define NB    2
#define NH    16
#define DH    64
#define DM    1024
#define MROWS (NB * SEQ)   // 4096

#if __has_builtin(__builtin_amdgcn_tensor_load_to_lds) && \
    __has_builtin(__builtin_amdgcn_s_wait_tensorcnt)
#define HAVE_TDM 1
#else
#define HAVE_TDM 0
#endif

#if __has_builtin(__builtin_amdgcn_permlane16)
#define HAVE_PERMLANE 1
#else
#define HAVE_PERMLANE 0
#endif

__device__ __forceinline__ __bf16 f2bf(float f) { return (__bf16)f; }

__device__ __forceinline__ v8f wmma_bf16(v16bf a, v16bf b, v8f c) {
  return __builtin_amdgcn_wmma_f32_16x16x32_bf16(false, a, false, b, (short)0, c,
                                                 false, false);
}

__device__ __forceinline__ v8f vzero8() {
  v8f z;
#pragma unroll
  for (int i = 0; i < 8; ++i) z[i] = 0.0f;
  return z;
}

// max-reduce across each 16-lane half of the wave (butterfly, pure VALU).
__device__ __forceinline__ float rowmax16(float x) {
#if HAVE_PERMLANE
  int v;
  v = __builtin_amdgcn_permlane16(__float_as_int(x), __float_as_int(x),
                                  (int)0x67452301, (int)0xEFCDAB89, false, false);
  x = fmaxf(x, __int_as_float(v));
  v = __builtin_amdgcn_permlane16(__float_as_int(x), __float_as_int(x),
                                  (int)0x54761032, (int)0xDCFE98BA, false, false);
  x = fmaxf(x, __int_as_float(v));
  v = __builtin_amdgcn_permlane16(__float_as_int(x), __float_as_int(x),
                                  (int)0x32107654, (int)0xBA98FEDC, false, false);
  x = fmaxf(x, __int_as_float(v));
  v = __builtin_amdgcn_permlane16(__float_as_int(x), __float_as_int(x),
                                  (int)0xFEDCBA98, (int)0x76543210, false, false);
  x = fmaxf(x, __int_as_float(v));
#else
#pragma unroll
  for (int off = 8; off; off >>= 1) x = fmaxf(x, __shfl_xor(x, off, 32));
#endif
  return x;
}

__device__ __forceinline__ unsigned lds_off(const void* p) {
  return (unsigned)(size_t)p;   // low 32 bits of generic LDS address = offset
}

#if HAVE_TDM
// 2-D TDM tile load, elem = 2B bf16. LDS rows padded by the DMA engine:
// (pad_amount+1) DWORDs inserted every 2<<pad_interval DWORDs.
__device__ __forceinline__ void tdm_load_2d(unsigned lds_addr,
                                            unsigned long long gaddr,
                                            unsigned tensor_d0, unsigned tensor_d1,
                                            unsigned long long stride0,
                                            unsigned tile_d0, unsigned tile_d1,
                                            unsigned pad_interval,
                                            unsigned pad_amount) {
  v4u g0;
  g0[0] = 1u;                                       // count=1, load, no gather
  g0[1] = lds_addr;                                 // [63:32] lds_addr
  g0[2] = (unsigned)gaddr;                          // [95:64] global_addr lo
  g0[3] = (unsigned)((gaddr >> 32) & 0x01ffffffu)   // [120:96] global_addr hi
          | (2u << 30);                             // [127:126] type=2 (image)
  v8i g1;
  g1[0] = (int)((1u << 16)                          // data_size = 2 bytes
                | (1u << 20)                        // pad_enable
                | (pad_interval << 22) | (pad_amount << 25));
  g1[1] = (int)(tensor_d0 << 16);                   // tensor_dim0 (lo 16)
  g1[2] = (int)((tensor_d0 >> 16) | (tensor_d1 << 16));
  g1[3] = (int)((tensor_d1 >> 16) | (tile_d0 << 16));
  g1[4] = (int)(tile_d1 & 0xffffu);                 // tile_dim1, tile_dim2=0
  g1[5] = (int)(unsigned)(stride0 & 0xffffffffull); // tensor_dim0_stride lo
  g1[6] = (int)(unsigned)((stride0 >> 32) & 0xffffull);
  g1[7] = 0;
  v4i z4; z4[0] = z4[1] = z4[2] = z4[3] = 0;
  v8i z8;
#pragma unroll
  for (int i = 0; i < 8; ++i) z8[i] = 0;
  __builtin_amdgcn_tensor_load_to_lds(g0, g1, z4, z4, z8, 0);
}
#endif

// ---------------------------------------------------------------------------
// fp32 -> bf16 elementwise convert
// ---------------------------------------------------------------------------
__global__ __launch_bounds__(256) void cvt_f32_bf16(const float* __restrict__ in,
                                                    __bf16* __restrict__ out, int n) {
  int i = blockIdx.x * 256 + threadIdx.x;
  if (i < n) out[i] = f2bf(in[i]);
}

// ---------------------------------------------------------------------------
// W [K][N] fp32 -> WT [N][K] bf16 (tiled transpose)
// ---------------------------------------------------------------------------
__global__ __launch_bounds__(256) void transpose_cvt(const float* __restrict__ W,
                                                     __bf16* __restrict__ WT,
                                                     int K, int N) {
  __shared__ float tile[32][33];
  const int tx = threadIdx.x & 31;
  const int ty = threadIdx.x >> 5;
  const int n0 = blockIdx.x * 32;
  const int k0 = blockIdx.y * 32;
#pragma unroll
  for (int i = 0; i < 32; i += 8)
    tile[ty + i][tx] = W[(size_t)(k0 + ty + i) * N + n0 + tx];
  __syncthreads();
#pragma unroll
  for (int i = 0; i < 32; i += 8)
    WT[(size_t)(n0 + ty + i) * K + k0 + tx] = f2bf(tile[tx][ty + i]);
}

// ---------------------------------------------------------------------------
// Generic bf16 WMMA GEMM: C[M,N] = A[M,K] * W[K,N] + bias[N]
//   A : bf16 [M][K], WT : bf16 [N][K]
// Block tile 128x64, 8 waves as 4(M)x2(N), wave tile 32x32, K-step 64
// (8 WMMAs per barrier). LDS double-buffered, filled by TDM.
// MODE 0: out bf16 [b,h,s,d]; MODE 1: out bf16 [b,h,d,s]; MODE 2: fp32 [M][N].
// ---------------------------------------------------------------------------
template <int MODE>
__global__ __launch_bounds__(256) void gemm_bf16_wmma(const __bf16* __restrict__ A,
                                                      const __bf16* __restrict__ WT,
                                                      const float* __restrict__ bias,
                                                      void* __restrict__ outp,
                                                      int M, int N, int K) {
  const int lane = threadIdx.x & 31;
  const int wave = threadIdx.x >> 5;
  const int hh   = lane >> 4;
  const int l16  = lane & 15;
  const int waveM = wave >> 1;
  const int waveN = wave & 1;
  const int m0 = blockIdx.y * 128;
  const int n0 = blockIdx.x * 64;

  __shared__ __bf16 As[2][128][72];   // 64 K-halves + 8 pad; row = 144B
  __shared__ __bf16 Bs[2][64][72];

  v8f acc[2][2];
#pragma unroll
  for (int mi = 0; mi < 2; ++mi)
#pragma unroll
    for (int ni = 0; ni < 2; ++ni) acc[mi][ni] = vzero8();

#if HAVE_TDM
  const unsigned long long Ag = (unsigned long long)(size_t)A;
  const unsigned long long Wg = (unsigned long long)(size_t)WT;
  if (threadIdx.x < 32) {  // wave 0 owns the DMA
    tdm_load_2d(lds_off(&As[0][0][0]), Ag + (size_t)m0 * K * 2,
                (unsigned)K, (unsigned)M, (unsigned)K, 64u, 128u, 4u, 3u);
    tdm_load_2d(lds_off(&Bs[0][0][0]), Wg + (size_t)n0 * K * 2,
                (unsigned)K, (unsigned)N, (unsigned)K, 64u, 64u, 4u, 3u);
  }
#endif

  int buf = 0;
  for (int k0 = 0; k0 < K; k0 += 64) {
#if HAVE_TDM
    if (threadIdx.x < 32) __builtin_amdgcn_s_wait_tensorcnt(0);
    __syncthreads();      // tile[buf] ready; tile[buf^1] readers (prev iter) done
    if (k0 + 64 < K && threadIdx.x < 32) {
      tdm_load_2d(lds_off(&As[buf ^ 1][0][0]),
                  Ag + ((size_t)m0 * K + k0 + 64) * 2,
                  (unsigned)K, (unsigned)M, (unsigned)K, 64u, 128u, 4u, 3u);
      tdm_load_2d(lds_off(&Bs[buf ^ 1][0][0]),
                  Wg + ((size_t)n0 * K + k0 + 64) * 2,
                  (unsigned)K, (unsigned)N, (unsigned)K, 64u, 64u, 4u, 3u);
    }
#else
    __syncthreads();
#pragma unroll
    for (int t = 0; t < 4; ++t) {       // A tile 128x64 = 1024 chunks of 8
      int chunk = threadIdx.x + t * 256;
      int row = chunk >> 3, c8 = chunk & 7;
      *(uint4*)&As[buf][row][c8 * 8] =
          *(const uint4*)(A + (size_t)(m0 + row) * K + k0 + c8 * 8);
    }
#pragma unroll
    for (int t = 0; t < 2; ++t) {       // B tile 64x64 = 512 chunks of 8
      int chunk = threadIdx.x + t * 256;
      int row = chunk >> 3, c8 = chunk & 7;
      *(uint4*)&Bs[buf][row][c8 * 8] =
          *(const uint4*)(WT + (size_t)(n0 + row) * K + k0 + c8 * 8);
    }
    __syncthreads();
#endif

    v16bf af[2][2], bf[2][2];
#pragma unroll
    for (int mi = 0; mi < 2; ++mi) {
      int row = waveM * 32 + mi * 16 + l16;
#pragma unroll
      for (int j = 0; j < 2; ++j) {
        v8bf lo = *(const v8bf*)&As[buf][row][j * 32 + 8 * hh];
        v8bf hi = *(const v8bf*)&As[buf][row][j * 32 + 16 + 8 * hh];
#pragma unroll
        for (int e = 0; e < 8; ++e) { af[mi][j][e] = lo[e]; af[mi][j][e + 8] = hi[e]; }
      }
    }
#pragma unroll
    for (int ni = 0; ni < 2; ++ni) {
      int row = waveN * 32 + ni * 16 + l16;
#pragma unroll
      for (int j = 0; j < 2; ++j) {
        const v8bf* p = (const v8bf*)&Bs[buf][row][j * 32 + 8 * hh];
        v8bf b0 = p[0], b1 = p[2];
#pragma unroll
        for (int e = 0; e < 8; ++e) { bf[ni][j][e] = b0[e]; bf[ni][j][e + 8] = b1[e]; }
      }
    }
#pragma unroll
    for (int j = 0; j < 2; ++j)
#pragma unroll
      for (int mi = 0; mi < 2; ++mi)
#pragma unroll
        for (int ni = 0; ni < 2; ++ni)
          acc[mi][ni] = wmma_bf16(af[mi][j], bf[ni][j], acc[mi][ni]);
    buf ^= 1;
  }

#pragma unroll
  for (int ni = 0; ni < 2; ++ni) {
    int n = n0 + waveN * 32 + ni * 16 + l16;
    float bv = bias ? bias[n] : 0.0f;
#pragma unroll
    for (int mi = 0; mi < 2; ++mi) {
#pragma unroll
      for (int r = 0; r < 8; ++r) {
        int m = m0 + waveM * 32 + mi * 16 + r + 8 * hh;
        float v = acc[mi][ni][r] + bv;
        if (MODE == 2) {
          ((float*)outp)[(size_t)m * N + n] = v;
        } else {
          int b = m >> 11, s = m & (SEQ - 1);
          int h = n >> 6, d = n & (DH - 1);
          __bf16 hv = f2bf(v);
          if (MODE == 0)
            ((__bf16*)outp)[(((size_t)b * NH + h) * SEQ + s) * DH + d] = hv;
          else
            ((__bf16*)outp)[(((size_t)b * NH + h) * DH + d) * SEQ + s] = hv;
        }
      }
    }
  }
}

// ---------------------------------------------------------------------------
// Flash attention: grid = B*H*(S/128), 256 threads (8 waves), wave = 16 q rows.
// K/V 64-row tiles double-buffered in LDS, filled by TDM.
// Row max: permlane16 butterfly (VALU). Row sum: P x ones WMMA (XDL).
// ---------------------------------------------------------------------------
__global__ __launch_bounds__(256) void attn_flash(const __bf16* __restrict__ Qg,
                                                  const __bf16* __restrict__ Kg,
                                                  const __bf16* __restrict__ VTg,
                                                  const float* __restrict__ mask,
                                                  __bf16* __restrict__ Ob) {
  const int lane = threadIdx.x & 31;
  const int wave = threadIdx.x >> 5;
  const int hh   = lane >> 4;
  const int l16  = lane & 15;

  int bid = blockIdx.x;
  const int qblk = bid & 15;  bid >>= 4;
  const int h    = bid & 15;  bid >>= 4;
  const int b    = bid;
  const int bh   = b * NH + h;
  const int q0   = qblk * 128 + wave * 16;

  __shared__ __bf16 Kt[2][64][72];   // [k][dk], row 144B
  __shared__ __bf16 Vt[2][64][72];   // [dv][k]
  __shared__ __bf16 Pl[8][16][72];   // per-wave P staging

  const __bf16* kbase = Kg  + (size_t)bh * SEQ * DH;
  const __bf16* vbase = VTg + (size_t)bh * DH * SEQ;

  // Q A-fragments held in registers for the whole loop
  v16bf qf[2];
  {
    const __bf16* qbase = Qg + ((size_t)bh * SEQ + q0 + l16) * DH;
#pragma unroll
    for (int j = 0; j < 2; ++j) {
      v8bf lo = *(const v8bf*)(qbase + j * 32 + 8 * hh);
      v8bf hi = *(const v8bf*)(qbase + j * 32 + 16 + 8 * hh);
#pragma unroll
      for (int e = 0; e < 8; ++e) { qf[j][e] = lo[e]; qf[j][e + 8] = hi[e]; }
    }
  }

  v16bf onesf;          // all-ones B fragment: row-sum of P via WMMA
#pragma unroll
  for (int e = 0; e < 16; ++e) onesf[e] = f2bf(1.0f);

  v8f o[4];
#pragma unroll
  for (int c = 0; c < 4; ++c) o[c] = vzero8();
  float m_run[8], l_run[8];
#pragma unroll
  for (int r = 0; r < 8; ++r) { m_run[r] = -3.0e38f; l_run[r] = 0.0f; }

  const float scale = 0.125f;  // 1/sqrt(64)

#if HAVE_TDM
  const unsigned long long kgb = (unsigned long long)(size_t)kbase;
  const unsigned long long vgb = (unsigned long long)(size_t)vbase;
  if (threadIdx.x < 32) {
    tdm_load_2d(lds_off(&Kt[0][0][0]), kgb, DH, SEQ, DH, 64u, 64u, 4u, 3u);
    tdm_load_2d(lds_off(&Vt[0][0][0]), vgb, SEQ, DH, SEQ, 64u, 64u, 4u, 3u);
  }
#endif

  int buf = 0;
  for (int k0 = 0; k0 < SEQ; k0 += 64) {
#if HAVE_TDM
    if (threadIdx.x < 32) __builtin_amdgcn_s_wait_tensorcnt(0);
    __syncthreads();   // tile[buf] ready; tile[buf^1] readers (prev iter) done
    if (k0 + 64 < SEQ && threadIdx.x < 32) {
      tdm_load_2d(lds_off(&Kt[buf ^ 1][0][0]),
                  kgb + (size_t)(k0 + 64) * DH * 2, DH, SEQ, DH, 64u, 64u, 4u, 3u);
      tdm_load_2d(lds_off(&Vt[buf ^ 1][0][0]),
                  vgb + (size_t)(k0 + 64) * 2, SEQ, DH, SEQ, 64u, 64u, 4u, 3u);
    }
#else
    __syncthreads();
#pragma unroll
    for (int t = 0; t < 2; ++t) {
      int chunk = threadIdx.x + t * 256;
      int row = chunk >> 3, c8 = chunk & 7;
      *(uint4*)&Kt[buf][row][c8 * 8] =
          *(const uint4*)(kbase + (size_t)(k0 + row) * DH + c8 * 8);
      *(uint4*)&Vt[buf][row][c8 * 8] =
          *(const uint4*)(vbase + (size_t)row * SEQ + k0 + c8 * 8);
    }
    __syncthreads();
#endif

    // ---- scores S = Q (16x64) . K^T ----
    v8f sc[4];
#pragma unroll
    for (int c = 0; c < 4; ++c) {
      v8f a = vzero8();
#pragma unroll
      for (int j = 0; j < 2; ++j) {
        const v8bf* p = (const v8bf*)&Kt[buf][c * 16 + l16][j * 32 + 8 * hh];
        v8bf b0 = p[0], b1 = p[2];
        v16bf bf;
#pragma unroll
        for (int e = 0; e < 8; ++e) { bf[e] = b0[e]; bf[e + 8] = b1[e]; }
        a = wmma_bf16(qf[j], bf, a);
      }
      sc[c] = a;
    }

    // scale + mask
#pragma unroll
    for (int c = 0; c < 4; ++c) {
      int kcol = k0 + c * 16 + l16;
#pragma unroll
      for (int r = 0; r < 8; ++r) {
        int qq = q0 + r + 8 * hh;
        sc[c][r] = sc[c][r] * scale + mask[(size_t)qq * SEQ + kcol];
      }
    }

    // ---- online softmax (row spread across 16 lanes) ----
    float alpha[8];
#pragma unroll
    for (int r = 0; r < 8; ++r) {
      float mx = fmaxf(fmaxf(sc[0][r], sc[1][r]), fmaxf(sc[2][r], sc[3][r]));
      mx = rowmax16(mx);
      float mn = fmaxf(m_run[r], mx);
      alpha[r] = __expf(m_run[r] - mn);
      m_run[r] = mn;
#pragma unroll
      for (int c = 0; c < 4; ++c) {
        float p = __expf(sc[c][r] - mn);
        o[c][r] *= alpha[r];
        Pl[wave][r + 8 * hh][c * 16 + l16] = f2bf(p);
      }
    }
    __syncthreads();   // P visible for A-fragment reload

    // ---- reload P as A fragments (hoisted: j only) ----
    v16bf paf[2];
#pragma unroll
    for (int j = 0; j < 2; ++j) {
      v8bf a0 = *(const v8bf*)&Pl[wave][l16][j * 32 + 8 * hh];
      v8bf a1 = *(const v8bf*)&Pl[wave][l16][j * 32 + 16 + 8 * hh];
#pragma unroll
      for (int e = 0; e < 8; ++e) { paf[j][e] = a0[e]; paf[j][e + 8] = a1[e]; }
    }

    // ---- row sum of P via WMMA with all-ones B ----
    {
      v8f lf = vzero8();
      lf = wmma_bf16(paf[0], onesf, lf);
      lf = wmma_bf16(paf[1], onesf, lf);
#pragma unroll
      for (int r = 0; r < 8; ++r) l_run[r] = l_run[r] * alpha[r] + lf[r];
    }

    // ---- O += P (16x64) . V ----
#pragma unroll
    for (int c = 0; c < 4; ++c) {
#pragma unroll
      for (int j = 0; j < 2; ++j) {
        const v8bf* p = (const v8bf*)&Vt[buf][c * 16 + l16][j * 32 + 8 * hh];
        v8bf b0 = p[0], b1 = p[2];
        v16bf bf;
#pragma unroll
        for (int e = 0; e < 8; ++e) { bf[e] = b0[e]; bf[e + 8] = b1[e]; }
        o[c] = wmma_bf16(paf[j], bf, o[c]);
      }
    }
    buf ^= 1;
  }

  // normalize + store O as bf16 [b, s, h*64+dv]
#pragma unroll
  for (int c = 0; c < 4; ++c)
#pragma unroll
    for (int r = 0; r < 8; ++r) {
      int qq = q0 + r + 8 * hh;
      float val = o[c][r] / l_run[r];
      Ob[((size_t)b * SEQ + qq) * DM + h * DH + c * 16 + l16] = f2bf(val);
    }
}

// ---------------------------------------------------------------------------
// Host-side orchestration. Workspace (bytes):
//   [0,8M) Xb | [8M,10M) WqT | [10M,12M) WkT | [12M,14M) WvT | [14M,16M) WoT
//   [16M,24M) Q [b,h,s,d] | [24M,32M) K | [32M,40M) VT [b,h,d,s] | [40M,48M) Ob
// ---------------------------------------------------------------------------
extern "C" void kernel_launch(void* const* d_in, const int* in_sizes, int n_in,
                              void* d_out, int out_size, void* d_ws, size_t ws_size,
                              hipStream_t stream) {
  const float* X    = (const float*)d_in[0];
  const float* mask = (const float*)d_in[1];
  const float* W_Q  = (const float*)d_in[2];
  const float* W_K  = (const float*)d_in[3];
  const float* W_V  = (const float*)d_in[4];
  const float* b_Q  = (const float*)d_in[5];
  const float* b_K  = (const float*)d_in[6];
  const float* b_V  = (const float*)d_in[7];
  const float* W_O  = (const float*)d_in[8];
  const float* b_O  = (const float*)d_in[9];

  char* ws = (char*)d_ws;
  const size_t MB = 1024u * 1024u;
  __bf16* Xb  = (__bf16*)(ws + 0);
  __bf16* WqT = (__bf16*)(ws + 8 * MB);
  __bf16* WkT = (__bf16*)(ws + 10 * MB);
  __bf16* WvT = (__bf16*)(ws + 12 * MB);
  __bf16* WoT = (__bf16*)(ws + 14 * MB);
  __bf16* Qg  = (__bf16*)(ws + 16 * MB);
  __bf16* Kg  = (__bf16*)(ws + 24 * MB);
  __bf16* VTg = (__bf16*)(ws + 32 * MB);
  __bf16* Ob  = (__bf16*)(ws + 40 * MB);

  {
    int n = MROWS * DM;
    cvt_f32_bf16<<<(n + 255) / 256, 256, 0, stream>>>(X, Xb, n);
    dim3 tg(DM / 32, DM / 32);
    transpose_cvt<<<tg, 256, 0, stream>>>(W_Q, WqT, DM, DM);
    transpose_cvt<<<tg, 256, 0, stream>>>(W_K, WkT, DM, DM);
    transpose_cvt<<<tg, 256, 0, stream>>>(W_V, WvT, DM, DM);
    transpose_cvt<<<tg, 256, 0, stream>>>(W_O, WoT, DM, DM);
  }

  dim3 ggrid(DM / 64, MROWS / 128);
  gemm_bf16_wmma<0><<<ggrid, 256, 0, stream>>>(Xb, WqT, b_Q, Qg,  MROWS, DM, DM);
  gemm_bf16_wmma<0><<<ggrid, 256, 0, stream>>>(Xb, WkT, b_K, Kg,  MROWS, DM, DM);
  gemm_bf16_wmma<1><<<ggrid, 256, 0, stream>>>(Xb, WvT, b_V, VTg, MROWS, DM, DM);

  attn_flash<<<NB * NH * (SEQ / 128), 256, 0, stream>>>(Qg, Kg, VTg, mask, Ob);

  gemm_bf16_wmma<2><<<ggrid, 256, 0, stream>>>(Ob, WoT, b_O, (float*)d_out,
                                               MROWS, DM, DM);
}
